// LawDNet_17763984736743
// MI455X (gfx1250) — matrix operand order repaired and checked by reading, not compiled, not run on hardware
//
#include <hip/hip_runtime.h>
#include <hip/hip_bf16.h>

#define B_SZ   4
#define C_SZ   256
#define H_SZ   64
#define W_SZ   64
#define P_SZ   256
#define NKP    5
#define G_SZ   60
#define PI_F   3.14159265358979323846f

typedef float    v2f  __attribute__((ext_vector_type(2)));
typedef float    v8f  __attribute__((ext_vector_type(8)));
typedef _Float16 v16h __attribute__((ext_vector_type(16)));

#if defined(__HIP_DEVICE_COMPILE__)

#if __has_builtin(__builtin_amdgcn_wmma_f32_16x16x4_f32)
#define USE_WMMA_F32X4 1
#else
#define USE_WMMA_F32X4 0
#endif

// One wave computes one 16x16 tile of  Apad[16x256] @ W[256 x ldb]  (cols col0..col0+15).
// A has B_SZ=4 valid rows; rows 4..15 are substituted with 0 in registers.
__device__ __forceinline__ v8f mlp_tile(const float* __restrict__ A,
                                        const float* __restrict__ Wm,
                                        int ldb, int col0, int lane)
{
  const int  lm   = lane & 15;
  const int  hi   = lane >> 4;
  const bool arow = (lm < B_SZ);
  const int  ar   = arow ? lm : 0;     // clamped row -> always-in-bounds load
  v8f acc = {};
#if USE_WMMA_F32X4
  // V_WMMA_F32_16X16X4_F32 fragment layout (ISA 7.12.2):
  //  A 16x4: lanes0-15 row=lane holds K={k0,k0+1}; lanes16-31 row=lane-16 holds K={k0+2,k0+3}
  //  B 4x16: lanes0-15 col=lane holds K={k0,k0+1}; lanes16-31 col=lane-16 holds K={k0+2,k0+3}
  for (int k0 = 0; k0 < P_SZ; k0 += 4) {
    const int ka = k0 + hi * 2;
    v2f a, b;
    const float a0 = A[ar * P_SZ + ka];
    const float a1 = A[ar * P_SZ + ka + 1];
    a.x = arow ? a0 : 0.f;
    a.y = arow ? a1 : 0.f;
    b.x = Wm[(ka)     * ldb + col0 + lm];
    b.y = Wm[(ka + 1) * ldb + col0 + lm];
    acc = __builtin_amdgcn_wmma_f32_16x16x4_f32(false, a, false, b,
                                                (short)0, acc, false, false);
  }
#else
  // Fallback: codegen-confirmed V_WMMA_F32_16X16X32_F16 (f32 accumulate).
  for (int k0 = 0; k0 < P_SZ; k0 += 32) {
    v16h a, b;
    #pragma unroll
    for (int e = 0; e < 8; ++e) {
      const int kA = k0 + hi * 8 + e;
      const float a0 = A[ar * P_SZ + kA];
      const float a1 = A[ar * P_SZ + kA + 16];
      a[e]     = arow ? (_Float16)a0 : (_Float16)0.f;
      a[e + 8] = arow ? (_Float16)a1 : (_Float16)0.f;
    }
    #pragma unroll
    for (int e = 0; e < 16; ++e) {
      const int kB = k0 + hi * 8 + e + (e >= 8 ? 8 : 0);
      b[e] = (_Float16)Wm[kB * ldb + col0 + lm];
    }
    acc = __builtin_amdgcn_wmma_f32_16x16x32_f16(false, a, false, b,
                                                 (short)0, acc, false, false);
  }
#endif
  return acc;
}
#endif // __HIP_DEVICE_COMPILE__

// pc = relu(para_code @ W_common + b_common)   [4 x 256], 16 waves, 1 block
__global__ void __launch_bounds__(512)
k_common(const float* __restrict__ para, const float* __restrict__ Wc,
         const float* __restrict__ bc, float* __restrict__ pc)
{
#if defined(__HIP_DEVICE_COMPILE__)
  const int lane = threadIdx.x & 31;
  const int nb   = (threadIdx.x >> 5) * 16;
  v8f acc = mlp_tile(para, Wc, P_SZ, nb, lane);
  const int lm = lane & 15, hi = lane >> 4;
  if (hi == 0) {                         // rows 0..7 live in the low half-wave
    #pragma unroll
    for (int r = 0; r < B_SZ; ++r) {
      const int n = nb + lm;
      pc[r * P_SZ + n] = fmaxf(acc[r] + bc[n], 0.f);
    }
  }
#endif
}

// Three heads fused: [kp(2560) | rad(1280) | sim(5120)] = 8960 cols = 560 tiles.
__global__ void __launch_bounds__(256)
k_heads(const float* __restrict__ pc,
        const float* __restrict__ Wkp,  const float* __restrict__ bkp,
        const float* __restrict__ Wrad, const float* __restrict__ brad,
        const float* __restrict__ Wsim, const float* __restrict__ bsim,
        float* __restrict__ hkp, float* __restrict__ hrad, float* __restrict__ hsim)
{
#if defined(__HIP_DEVICE_COMPILE__)
  const int lane = threadIdx.x & 31;
  const int wid  = blockIdx.x * (blockDim.x >> 5) + (threadIdx.x >> 5);
  const int ng   = wid * 16;
  const float* Wm; const float* bias; float* outp;
  int ldb, col0, act;                    // act: 0 tanh, 1 sigmoid
  if (ng < 2 * NKP * C_SZ) {                                    // kp
    Wm = Wkp;  bias = bkp;  outp = hkp;  ldb = 2 * NKP * C_SZ; col0 = ng;                    act = 0;
  } else if (ng < 3 * NKP * C_SZ) {                             // rad
    Wm = Wrad; bias = brad; outp = hrad; ldb = NKP * C_SZ;     col0 = ng - 2 * NKP * C_SZ;  act = 1;
  } else {                                                      // sim
    Wm = Wsim; bias = bsim; outp = hsim; ldb = 4 * NKP * C_SZ; col0 = ng - 3 * NKP * C_SZ;  act = 0;
  }
  v8f acc = mlp_tile(pc, Wm, ldb, col0, lane);
  const int lm = lane & 15, hi = lane >> 4;
  if (hi == 0) {
    #pragma unroll
    for (int r = 0; r < B_SZ; ++r) {
      const int n = col0 + lm;
      float v = acc[r] + bias[n];
      v = act ? (1.f / (1.f + __expf(-v))) : tanhf(v);
      outp[r * ldb + n] = v;
    }
  }
#endif
}

// Per-(b,c) block: 5-keypoint warp field + softmax blend; write straight into the
// torch "reshape dance" layout rs[B][2C][60][60]:
//   c2 = 2c + (j>=30), g1 = 2(j%30) + (i>=30), g2 = 2(i%30) + k
__global__ void __launch_bounds__(256)
k_field(const float* __restrict__ hkp, const float* __restrict__ hrad,
        const float* __restrict__ hsim, float* __restrict__ rs)
{
#if defined(__HIP_DEVICE_COMPILE__)
  const int bc = blockIdx.x;
  const int b = bc >> 8, c = bc & (C_SZ - 1);
  float kx[NKP], ky[NKP], irr[NKP], s0[NKP], s1[NKP], s2[NKP], s3[NKP];
  #pragma unroll
  for (int n = 0; n < NKP; ++n) {
    const int nc = n * C_SZ + c;
    kx[n]  = hkp[b * (2 * NKP * C_SZ) + nc * 2 + 0];
    ky[n]  = hkp[b * (2 * NKP * C_SZ) + nc * 2 + 1];
    irr[n] = -1.f / hrad[b * (NKP * C_SZ) + nc];
    const float* sp = &hsim[b * (4 * NKP * C_SZ) + nc * 4];
    s0[n] = sp[0]; s1[n] = sp[1]; s2[n] = sp[2]; s3[n] = sp[3];
  }
  for (int p = threadIdx.x; p < G_SZ * G_SZ; p += blockDim.x) {
    const int i = p / G_SZ;              // row -> y
    const int j = p - i * G_SZ;          // col -> x
    const float xj = -1.f + (2.f / (G_SZ - 1)) * (float)j;
    const float yi = -1.f + (2.f / (G_SZ - 1)) * (float)i;
    float e[NKP], gx[NKP], gy[NKP];
    float emax = -1e30f;
    #pragma unroll
    for (int n = 0; n < NKP; ++n) {
      const float ox = xj - kx[n];
      const float oy = yi - ky[n];
      const float en = __expf(sqrtf(ox * ox + oy * oy) * irr[n]);
      e[n] = en; emax = fmaxf(emax, en);
      const float scale = 1.f + s0[n] * en;
      float sa, ca;
      __sincosf(s1[n] * PI_F * en, &sa, &ca);
      gx[n] = scale * ( ox * ca + oy * sa) + s2[n] * en + kx[n];
      gy[n] = scale * (-ox * sa + oy * ca) + s3[n] * en + ky[n];
    }
    float wsum = 0.f, ax = 0.f, ay = 0.f;
    #pragma unroll
    for (int n = 0; n < NKP; ++n) {
      const float wn = __expf(e[n] - emax);    // softmax over keypoints
      wsum += wn; ax += wn * gx[n]; ay += wn * gy[n];
    }
    const float inv = 1.f / wsum;
    const int c2 = 2 * c + (j >= 30);
    const int g1 = 2 * (j % 30) + (i >= 30);
    const int g2 = 2 * (i % 30);
    float* dst = &rs[(((size_t)b * 2 * C_SZ + c2) * G_SZ + g1) * G_SZ + g2];
    dst[0] = ax * inv;
    dst[1] = ay * inv;
  }
#endif
}

// Fused: bilinear resize 60->64 (half-pixel, clamped == jax renormalized edges)
// + grid_sample. depth plane is exactly z=c, so trilinear collapses to 2D bilinear.
__global__ void __launch_bounds__(256)
k_sample(const float* __restrict__ rs, const float* __restrict__ fm,
         float* __restrict__ out)
{
#if defined(__HIP_DEVICE_COMPILE__)
  const int idx = blockIdx.x * blockDim.x + threadIdx.x;
  if (idx >= B_SZ * C_SZ * H_SZ * W_SZ) return;
  const int w = idx & (W_SZ - 1);
  const int h = (idx >> 6) & (H_SZ - 1);
  const int c = (idx >> 12) & (C_SZ - 1);
  const int b = idx >> 20;

  const float sy = (h + 0.5f) * ((float)G_SZ / H_SZ) - 0.5f;
  const float sx = (w + 0.5f) * ((float)G_SZ / W_SZ) - 0.5f;
  const float fy = floorf(sy), fx = floorf(sx);
  const float ty = sy - fy,   tx = sx - fx;
  const int y0 = min(max((int)fy, 0), G_SZ - 1);
  const int y1 = min(max((int)fy + 1, 0), G_SZ - 1);
  const int x0 = min(max((int)fx, 0), G_SZ - 1);
  const int x1 = min(max((int)fx + 1, 0), G_SZ - 1);

  const float* plane = &fm[((size_t)b * C_SZ + c) * (H_SZ * W_SZ)];
  __builtin_prefetch(plane, 0, 0);     // global_prefetch_b8

  const float* chx = &rs[((size_t)b * 2 * C_SZ + 2 * c) * (G_SZ * G_SZ)];
  const float* chy = chx + G_SZ * G_SZ;
  const float w00 = (1.f - ty) * (1.f - tx), w01 = (1.f - ty) * tx;
  const float w10 = ty * (1.f - tx),         w11 = ty * tx;
  const float gx = w00 * chx[y0 * G_SZ + x0] + w01 * chx[y0 * G_SZ + x1]
                 + w10 * chx[y1 * G_SZ + x0] + w11 * chx[y1 * G_SZ + x1];
  const float gy = w00 * chy[y0 * G_SZ + x0] + w01 * chy[y0 * G_SZ + x1]
                 + w10 * chy[y1 * G_SZ + x0] + w11 * chy[y1 * G_SZ + x1];

  const float ix = ((gx + 1.f) * W_SZ - 1.f) * 0.5f;
  const float iy = ((gy + 1.f) * H_SZ - 1.f) * 0.5f;
  const float fxi = floorf(ix), fyi = floorf(iy);
  const float txi = ix - fxi,   tyi = iy - fyi;
  const int sx0 = (int)fxi, sy0 = (int)fyi;
  float acc = 0.f;
  #pragma unroll
  for (int dy = 0; dy < 2; ++dy) {
    const int yy = sy0 + dy;
    if (yy < 0 || yy >= H_SZ) continue;
    const float wy = dy ? tyi : (1.f - tyi);
    #pragma unroll
    for (int dx = 0; dx < 2; ++dx) {
      const int xx = sx0 + dx;
      if (xx < 0 || xx >= W_SZ) continue;
      const float wx = dx ? txi : (1.f - txi);
      acc += wy * wx * plane[yy * W_SZ + xx];
    }
  }
  out[idx] = acc;
#endif
}

extern "C" void kernel_launch(void* const* d_in, const int* in_sizes, int n_in,
                              void* d_out, int out_size, void* d_ws, size_t ws_size,
                              hipStream_t stream) {
  const float* fm   = (const float*)d_in[0];
  const float* para = (const float*)d_in[1];
  const float* Wc   = (const float*)d_in[2];
  const float* bc   = (const float*)d_in[3];
  const float* Wkp  = (const float*)d_in[4];
  const float* bkp  = (const float*)d_in[5];
  const float* Wrad = (const float*)d_in[6];
  const float* brad = (const float*)d_in[7];
  const float* Wsim = (const float*)d_in[8];
  const float* bsim = (const float*)d_in[9];
  float* out = (float*)d_out;

  char* ws = (char*)d_ws;
  float* pc   = (float*)(ws + 0);        //  4*256*4        =  4 KB
  float* hkp  = (float*)(ws + 4096);     //  4*2560*4       = 40 KB
  float* hrad = (float*)(ws + 45056);    //  4*1280*4       = 20 KB
  float* hsim = (float*)(ws + 65536);    //  4*5120*4       = 80 KB
  float* rs   = (float*)(ws + 147456);   //  4*512*3600*4   = 29.5 MB (L2-resident)

  k_common<<<1, 512, 0, stream>>>(para, Wc, bc, pc);
  k_heads <<<70, 256, 0, stream>>>(pc, Wkp, bkp, Wrad, brad, Wsim, bsim, hkp, hrad, hsim);
  k_field <<<B_SZ * C_SZ, 256, 0, stream>>>(hkp, hrad, hsim, rs);
  const int total = B_SZ * C_SZ * H_SZ * W_SZ;
  k_sample<<<(total + 255) / 256, 256, 0, stream>>>(rs, fm, out);

  (void)in_sizes; (void)n_in; (void)out_size; (void)ws_size;
}
